// DyResBlock_36679020708038
// MI455X (gfx1250) — matrix operand order repaired and checked by reading, not compiled, not run on hardware
//
#include <hip/hip_runtime.h>

typedef __attribute__((ext_vector_type(16))) __bf16          v16bf;
typedef __attribute__((ext_vector_type(8)))  float           v8f;
typedef __attribute__((ext_vector_type(8)))  unsigned short  ushort8;

#define NB   4
#define C128 128
#define HW   16384          // 128*128
#define NPIX 65536          // 4*128*128
#define NELT 8388608        // 4*128*128*128

static __device__ __forceinline__ unsigned short f2bf(float f) {
    unsigned int u = __float_as_uint(f);
    unsigned int r = (u + 0x7FFFu + ((u >> 16) & 1u)) >> 16;   // RNE
    return (unsigned short)r;
}

// ---------------------------------------------------------------- mask pass
__global__ void zero_counts_kernel(int* counts) {
    if (threadIdx.x < 12) counts[threadIdx.x] = 0;
}

__global__ __launch_bounds__(256) void mask_kernel(
    const float* __restrict__ x, const float* __restrict__ gum,
    const float* __restrict__ mask_w, unsigned char* __restrict__ branch,
    int* __restrict__ counts)
{
    int p = blockIdx.x * 256 + threadIdx.x;           // 0..NPIX-1
    if (p >= NPIX) return;
    int n = p >> 14;
    int hw = p & (HW - 1);
    const float* xp = x + (size_t)n * C128 * HW + hw;
    float l0 = 0.f, l1 = 0.f, l2 = 0.f;
    for (int c = 0; c < C128; ++c) {
        float v = xp[(size_t)c * HW];
        l0 += v * mask_w[c];
        l1 += v * mask_w[C128 + c];
        l2 += v * mask_w[2 * C128 + c];
    }
    l0 += gum[((size_t)n * 3 + 0) * HW + hw];
    l1 += gum[((size_t)n * 3 + 1) * HW + hw];
    l2 += gum[((size_t)n * 3 + 2) * HW + hw];
    int idx = 0; float best = l0;
    if (l1 > best) { best = l1; idx = 1; }
    if (l2 > best) { best = l2; idx = 2; }
    branch[p] = (unsigned char)idx;
    atomicAdd(&counts[n * 3 + idx], 1);
}

__global__ void sparsity_kernel(const int* __restrict__ counts, float* __restrict__ sp) {
    int n = threadIdx.x;
    if (n < NB) {
        float s = 0.0633f * (float)counts[n * 3 + 0]
                + 0.5555f * (float)counts[n * 3 + 1]
                +           (float)counts[n * 3 + 2];
        sp[n] = s / (float)HW;
    }
}

// ---------------------------------------------------------------- convert x
__global__ __launch_bounds__(256) void convert_x_kernel(
    const float* __restrict__ x, unsigned short* __restrict__ xbf)
{
    size_t e = (size_t)blockIdx.x * 256 + threadIdx.x;
    if (e < NELT) xbf[e] = f2bf(x[e]);
}

// --------------------------------------------------- WMMA implicit-GEMM conv
// C[n, m, h, w] (+)= sum_{tap, ci} W[m, e(ci,tap)] * In[n, ci, h+dh, w+dw]
// IS3x3: K = 1152 (e = ci*9 + tap); else 1x1: K = 128 (e = ci).
// Block: 256 threads (8 waves). Tile: 128 M x 64 N (one (n,h) row, half width).
// Activation halo (3 rows x 66 cols x 32 ch) staged in LDS once per 32-channel
// block; all 9 taps run out of LDS (9x less HBM traffic than naive im2col).
template <bool IS3X3>
__global__ __launch_bounds__(256, 2) void gemm_conv_wmma(
    const unsigned short* __restrict__ In,   // bf16 NCHW
    const float* __restrict__ W,             // f32 (128 x K)
    float* __restrict__ C,                   // f32 NCHW
    int accumulate)
{
    constexpr int HALO_ROWS = IS3X3 ? 3 : 1;
    constexpr int HALO_W    = IS3X3 ? 66 : 64;
    constexpr int TAPS      = IS3X3 ? 9 : 1;
    constexpr int KTOT      = IS3X3 ? 1152 : 128;
    constexpr int HALO_ELEMS = HALO_ROWS * 32 * HALO_W;
    constexpr int BITER = (HALO_ELEMS + 255) / 256;

    __shared__ __align__(16) unsigned short ldsA[C128 * 40];               // [m][k]
    __shared__ __align__(16) unsigned short ldsB[HALO_ROWS * HALO_W * 40]; // [row][pix][ci]

    int bid = blockIdx.x;                  // 0..1023
    int n   = bid >> 8;
    int rem = bid & 255;
    int hh  = rem >> 1;
    int w0  = (rem & 1) * 64;

    int tid  = threadIdx.x;
    int wave = tid >> 5;
    int lane = tid & 31;
    int lrow = lane & 15;
    int g    = lane >> 4;

    v8f acc[4] = {};

    for (int cblk4 = 0; cblk4 < 4; ++cblk4) {
        int cblk = cblk4 << 5;

        // ---- stage activation halo: HALO_ROWS x 32ci x HALO_W, zero-padded,
        //      transposed to [row][pix][ci] so B fragments are K-contiguous ----
        #pragma unroll
        for (int r = 0; r < BITER; ++r) {
            int lin = tid + (r << 8);
            if (lin < HALO_ELEMS) {
                int wpos = lin % HALO_W;
                int t2   = lin / HALO_W;
                int ci   = t2 & 31;
                int row  = t2 >> 5;
                int hrow = IS3X3 ? (hh + row - 1) : hh;
                int gw   = IS3X3 ? (w0 + wpos - 1) : (w0 + wpos);
                unsigned short v = 0;
                if (hrow >= 0 && hrow < 128 && gw >= 0 && gw < 128)
                    v = In[(((size_t)(n * C128 + cblk + ci)) * 128 + hrow) * 128 + gw];
                ldsB[(row * HALO_W + wpos) * 40 + ci] = v;
            }
        }
        // ---- prefetch next channel block's rows (gfx1250 global_prefetch_b8) ----
        if (IS3X3 && cblk4 < 3 && tid < 32 * HALO_ROWS) {
            int ci2  = cblk + 32 + (tid & 31);
            int row2 = hh + (tid >> 5) - 1;
            if (row2 >= 0 && row2 < 128)
                __builtin_prefetch(
                    &In[(((size_t)(n * C128 + ci2)) * 128 + row2) * 128 + w0], 0, 1);
        }

        #pragma unroll
        for (int tap = 0; tap < TAPS; ++tap) {
            // ---- stage A tile: 128 x 32 weights (f32 -> bf16) ----
            #pragma unroll
            for (int r = 0; r < 16; ++r) {
                int lin = tid + (r << 8);           // 0..4095
                int m = lin >> 5, j = lin & 31;
                int ci = cblk + j;
                int e = IS3X3 ? (ci * 9 + tap) : ci;
                ldsA[m * 40 + j] = f2bf(W[(size_t)m * KTOT + e]);
            }
            __syncthreads();   // halo + A ready

            // ---- fragments + WMMA ----
            // A 16x32 bf16: lane(l<16): k 0..7 & 16..23; lane(l>=16): k 8..15 & 24..31
            union Frag { ushort8 h[2]; v16bf v; };
            int arow = (wave << 4) + lrow;
            Frag fa;
            fa.h[0] = *(const ushort8*)&ldsA[arow * 40 + g * 8];
            fa.h[1] = *(const ushort8*)&ldsA[arow * 40 + 16 + g * 8];

            int rowIdx = IS3X3 ? (tap / 3) : 0;     // dh + 1
            int dwi    = IS3X3 ? (tap % 3) : 0;     // dw + 1
            #pragma unroll
            for (int nt = 0; nt < 4; ++nt) {
                // B 32x16 bf16: lanes 0-15 hold k 0..7 & 16..23, lanes 16-31 k 8..15 & 24..31; col = lane%16
                int wcol = (nt << 4) + lrow + dwi;
                const unsigned short* bp = &ldsB[(rowIdx * HALO_W + wcol) * 40];
                Frag fb;
                fb.h[0] = *(const ushort8*)&bp[(g << 3)];
                fb.h[1] = *(const ushort8*)&bp[16 + (g << 3)];
                acc[nt] = __builtin_amdgcn_wmma_f32_16x16x32_bf16(
                    false, fa.v, false, fb.v, (short)0, acc[nt], false, false);
            }
            __syncthreads();   // protect ldsA (next tap) / ldsB (next cblk)
        }
    }

    // ---- store: C/D layout: VGPR r of lane L -> M = r + 8*(L/16), N = L%16 ----
    #pragma unroll
    for (int nt = 0; nt < 4; ++nt) {
        int gw = w0 + (nt << 4) + lrow;
        #pragma unroll
        for (int r = 0; r < 8; ++r) {
            int m = (wave << 4) + (g << 3) + r;
            size_t o = (((size_t)(n * C128 + m)) * 128 + hh) * 128 + gw;
            float v = acc[nt][r];
            if (accumulate) C[o] += v; else C[o] = v;
        }
    }
}

// ----------------------------------------------------------- branch prepare
__global__ __launch_bounds__(256) void branch_prep_kernel(
    const float* __restrict__ x, const float* __restrict__ mid0_raw,
    const float* __restrict__ hig0_raw, const unsigned char* __restrict__ branch,
    const float* __restrict__ low_w,
    unsigned short* __restrict__ lowb, unsigned short* __restrict__ midb,
    unsigned short* __restrict__ higb)
{
    size_t e = (size_t)blockIdx.x * 256 + threadIdx.x;
    if (e >= NELT) return;
    int wq = (int)(e & 127);
    size_t t = e >> 7;
    int hq = (int)(t & 127); t >>= 7;
    int c  = (int)(t & 127);
    int n  = (int)(t >> 7);
    int p  = (n << 14) | (hq << 7) | wq;
    int idx = branch[p];
    unsigned short lo = 0, mi = 0, hi = 0;
    if (idx == 0) {
        float s = 0.f;
        #pragma unroll
        for (int kh = 0; kh < 3; ++kh) {
            int hh2 = hq + kh - 1;
            if (hh2 < 0 || hh2 > 127) continue;
            #pragma unroll
            for (int kw = 0; kw < 3; ++kw) {
                int ww = wq + kw - 1;
                if (ww < 0 || ww > 127) continue;
                s += x[(((size_t)(n * C128 + c)) * 128 + hh2) * 128 + ww] * low_w[c * 9 + kh * 3 + kw];
            }
        }
        lo = f2bf(s > 0.f ? s : 0.f);
    } else if (idx == 1) {
        float s = mid0_raw[e];
        mi = f2bf(s > 0.f ? s : 0.f);
    } else {
        float s = hig0_raw[e];
        hi = f2bf(s > 0.f ? s : 0.f);
    }
    lowb[e] = lo; midb[e] = mi; higb[e] = hi;
}

// ------------------------------------------------------------------ finalize
__global__ __launch_bounds__(256) void finalize_kernel(
    const float* __restrict__ x, const float* __restrict__ acc_lowmid,
    const float* __restrict__ hig_raw, const unsigned char* __restrict__ branch,
    float* __restrict__ out)
{
    size_t e = (size_t)blockIdx.x * 256 + threadIdx.x;
    if (e >= NELT) return;
    int wq = (int)(e & 127);
    size_t t = e >> 7;
    int hq = (int)(t & 127); t >>= 7;
    int n  = (int)(t >> 7);
    int p  = (n << 14) | (hq << 7) | wq;
    float v = x[e] + acc_lowmid[e];
    if (branch[p] == 2) v += hig_raw[e];
    out[e] = v;
}

// ------------------------------------------------------------------- launch
extern "C" void kernel_launch(void* const* d_in, const int* in_sizes, int n_in,
                              void* d_out, int out_size, void* d_ws, size_t ws_size,
                              hipStream_t stream)
{
    (void)in_sizes; (void)n_in; (void)out_size; (void)ws_size;
    const float* x      = (const float*)d_in[0];
    const float* gum    = (const float*)d_in[1];
    const float* mask_w = (const float*)d_in[2];
    const float* low_w  = (const float*)d_in[3];
    const float* mid_w  = (const float*)d_in[4];
    const float* hig_w0 = (const float*)d_in[5];
    const float* hig_w1 = (const float*)d_in[6];
    const float* w1_low = (const float*)d_in[7];
    const float* w1_mid = (const float*)d_in[8];
    float* out = (float*)d_out;
    float* sp  = out + (size_t)NELT;         // 4 sparsity floats

    char* ws = (char*)d_ws;
    unsigned char* branch = (unsigned char*)ws;                 // 65536 B
    int* counts = (int*)(ws + 65536);                           // 48 B
    size_t off = 65536 + 256;
    unsigned short* xbf  = (unsigned short*)(ws + off); off += (size_t)NELT * 2;
    unsigned short* lowb = (unsigned short*)(ws + off); off += (size_t)NELT * 2;
    unsigned short* midb = (unsigned short*)(ws + off); off += (size_t)NELT * 2;
    unsigned short* higb = (unsigned short*)(ws + off); off += (size_t)NELT * 2;
    float* buf0 = (float*)(ws + off); off += (size_t)NELT * 4;  // mid0_raw -> low/mid acc
    float* buf1 = (float*)(ws + off);                           // hig0_raw -> hig_raw

    const int eltBlocks = NELT / 256;        // 32768
    const int gemmBlocks = NB * 128 * 2;     // 1024

    zero_counts_kernel<<<1, 32, 0, stream>>>(counts);
    mask_kernel<<<NPIX / 256, 256, 0, stream>>>(x, gum, mask_w, branch, counts);
    sparsity_kernel<<<1, 32, 0, stream>>>(counts, sp);
    convert_x_kernel<<<eltBlocks, 256, 0, stream>>>(x, xbf);

    // mid0_raw = conv3x3(x, mid_w); hig0_raw = conv3x3(x, hig_w0)
    gemm_conv_wmma<true><<<gemmBlocks, 256, 0, stream>>>(xbf, mid_w, buf0, 0);
    gemm_conv_wmma<true><<<gemmBlocks, 256, 0, stream>>>(xbf, hig_w0, buf1, 0);

    branch_prep_kernel<<<eltBlocks, 256, 0, stream>>>(x, buf0, buf1, branch, low_w,
                                                      lowb, midb, higb);

    // hig_raw = conv3x3(higb, hig_w1)  (overwrites buf1; input already extracted)
    gemm_conv_wmma<true><<<gemmBlocks, 256, 0, stream>>>(higb, hig_w1, buf1, 0);
    // buf0 = w1_low @ lowb ; buf0 += w1_mid @ midb  (1x1 GEMMs)
    gemm_conv_wmma<false><<<gemmBlocks, 256, 0, stream>>>(lowb, w1_low, buf0, 0);
    gemm_conv_wmma<false><<<gemmBlocks, 256, 0, stream>>>(midb, w1_mid, buf0, 1);

    finalize_kernel<<<eltBlocks, 256, 0, stream>>>(x, buf0, buf1, branch, out);
}
